// CSSMBlock_37950331027901
// MI455X (gfx1250) — matrix operand order repaired and checked by reading, not compiled, not run on hardware
//
#include <hip/hip_runtime.h>
#include <hip/hip_bf16.h>

typedef __attribute__((ext_vector_type(16))) _Float16 v16h;
typedef __attribute__((ext_vector_type(8)))  _Float16 v8h;
typedef __attribute__((ext_vector_type(8)))  float    v8f;
typedef unsigned int u32x4 __attribute__((ext_vector_type(4)));
typedef unsigned int u32x8 __attribute__((ext_vector_type(8)));

#define DIMC    64
#define D_INNER 128
#define HWSZ    16384          // 128*128
#define NB      2
#define NTOK    32768          // NB * HWSZ
#define PW      130            // padded width/height (1-px halo)
#define PROW    (PW * 64)      // NHWC padded row: 130 px * 64 ch = 8320 halfs
#define PIMG    (PW * PROW)    // one padded image  = 1,081,600 halfs

__device__ __forceinline__ v16h cat8(v8h lo, v8h hi) {
  return __builtin_shufflevector(lo, hi, 0, 1, 2, 3, 4, 5, 6, 7,
                                 8, 9, 10, 11, 12, 13, 14, 15);
}

// ---------------------------------------------------------------------------
// WMMA 16x16 tile helper: A is 16 x K row-major (lda), B is W[n][k] row-major
// (ldb).  K multiple of 32; lda/ldb multiples of 8 (16B-aligned rows).
// ---------------------------------------------------------------------------
__device__ __forceinline__ v8f wmma_tile_f16(const _Float16* __restrict__ A, int lda,
                                             const _Float16* __restrict__ Bw, int ldb,
                                             int K, v8f acc) {
  const int lane   = threadIdx.x & 31;
  const int r      = lane & 15;
  const int halfid = lane >> 4;
  const _Float16* arow = A  + (size_t)r * lda;
  const _Float16* brow = Bw + (size_t)r * ldb;
  for (int k0 = 0; k0 < K; k0 += 32) {
    v8h alo = *(const v8h*)(arow + k0 + halfid * 16);
    v8h ahi = *(const v8h*)(arow + k0 + halfid * 16 + 8);
    v8h blo = *(const v8h*)(brow + k0 + halfid * 16);
    v8h bhi = *(const v8h*)(brow + k0 + halfid * 16 + 8);
    acc = __builtin_amdgcn_wmma_f32_16x16x32_f16(false, cat8(alo, ahi),
                                                 false, cat8(blo, bhi),
                                                 (short)0, acc, false, false);
  }
  return acc;
}

// K00: zero both padded f16 activation buffers (halo ring must be zero)
__global__ void zero_padded(_Float16* __restrict__ p1, _Float16* __restrict__ p2, int n) {
  int i = blockIdx.x * blockDim.x + threadIdx.x;
  if (i < n) { p1[i] = (_Float16)0.f; p2[i] = (_Float16)0.f; }
}

// K0a: pack / convert all weights once.
__global__ void pack_weights(const float* __restrict__ c1w, const float* __restrict__ c2w,
                             const float* __restrict__ ipw, const float* __restrict__ xpw,
                             const float* __restrict__ opw, const float* __restrict__ alog,
                             _Float16* __restrict__ wt1, _Float16* __restrict__ wt2,
                             _Float16* __restrict__ inw, _Float16* __restrict__ xpwh,
                             _Float16* __restrict__ outw, float* __restrict__ Aneg) {
  int j = blockIdx.x * blockDim.x + threadIdx.x;
  if (j < 9 * 64 * 64) {
    int tap = j >> 12, co = (j >> 6) & 63, ci = j & 63;
    int kh = tap / 3, kw = tap % 3;
    int src = ((co * 64 + ci) * 3 + kh) * 3 + kw;
    wt1[j] = (_Float16)c1w[src];
    wt2[j] = (_Float16)c2w[src];
    return;
  }
  j -= 9 * 64 * 64;
  if (j < 256 * 64) { inw[j] = (_Float16)ipw[j]; return; }
  j -= 256 * 64;
  if (j < 48 * 128) {
    int row = j >> 7, col = j & 127;
    xpwh[j] = (_Float16)(row < 36 ? xpw[row * 128 + col] : 0.f);
    return;
  }
  j -= 48 * 128;
  if (j < 64 * 128) { outw[j] = (_Float16)opw[j]; return; }
  j -= 64 * 128;
  if (j < 128 * 16) { Aneg[j] = -__expf(alog[j]); return; }
}

// K0b: x (NCHW f32) -> padded NHWC f16 [b][y+1][x+1][ci]
__global__ void cast_to_padded(const float* __restrict__ in, _Float16* __restrict__ out) {
  int idx = blockIdx.x * blockDim.x + threadIdx.x;
  if (idx >= NB * 64 * HWSZ) return;
  int cib = idx >> 14, l = idx & 16383, y = l >> 7, xx = l & 127;
  int b = cib >> 6, ci = cib & 63;
  out[(size_t)b * PIMG + (size_t)(y + 1) * PROW + (size_t)(xx + 1) * 64 + ci]
      = (_Float16)in[idx];
}

// ---------------------------------------------------------------------------
// K1/K2: 3x3 conv 64->64, implicit GEMM on WMMA, NHWC activations.
// One block = one pixel tile (16 contiguous w), 4 waves = 4 co tiles.
// Halo tile = 3 rows of (18 px x 64 ch); each row is a CONTIGUOUS 2304-byte
// run in NHWC, DMA'd to LDS by the Tensor Data Mover (waves 0..2, one row
// each).  WMMA B operands then come from LDS as contiguous ds_load_b128.
// mode 0: +bias, ReLU -> padded NHWC f16.  mode 1: +bias, +residual -> f32.
// ---------------------------------------------------------------------------
__global__ void conv3x3_wmma(const _Float16* __restrict__ actp,  // padded NHWC
                             const _Float16* __restrict__ wt,    // [9][64][64]
                             const float* __restrict__ bias,
                             const float* __restrict__ resid,
                             float* __restrict__ out_f,
                             _Float16* __restrict__ out_hp,
                             int mode) {
  __shared__ __align__(16) _Float16 smem[3 * 18 * 64];  // [y][x][ci], ci fastest
  const int lane   = threadIdx.x & 31;
  const int wv     = threadIdx.x >> 5;
  const int pt     = blockIdx.x;
  const int b      = pt >> 10;
  const int rem    = pt & 1023;
  const int h      = rem >> 3;
  const int w0     = (rem & 7) << 4;
  const int n      = lane & 15;
  const int halfid = lane >> 4;
  const int w      = w0 + n;
  const int co0    = wv * 16;

  // --- TDM: waves 0..2 each copy one contiguous halo row (1152 halfs)
  if (wv < 3) {
    const unsigned long long gaddr = (unsigned long long)(const char*)actp
        + 2ull * ((unsigned long long)b * PIMG
                  + (unsigned long long)(h + wv) * PROW      // padded y = h-1+wv+1
                  + (unsigned long long)w0 * 64ull);         // padded x = w0-1+1
    const unsigned ldsa = (unsigned)(size_t)&smem[wv * 18 * 64];
    u32x4 g0; u32x8 g1; u32x4 g2; u32x4 g3;
    g0[0] = 1u;                                       // count = 1 valid D#
    g0[1] = ldsa;                                     // lds_addr
    g0[2] = (unsigned)gaddr;                          // global_addr[31:0]
    g0[3] = (unsigned)((gaddr >> 32) & 0x1FFFFFFull)  // global_addr[56:32]
          | (2u << 30);                               // type = 2 (image)
    g1[0] = 1u << 16;                                 // data_size = 2 bytes
    g1[1] = 1152u << 16;                              // tensor_dim0 = 1152
    g1[2] = 0u;                                       // (dim0 hi, dim1 lo)
    g1[3] = 1152u << 16;                              // tile_dim0 = 1152
    g1[4] = 0u;                                       // tile_dim1/2 unused
    g1[5] = 1152u;                                    // tensor_dim0_stride
    g1[6] = 0u;
    g1[7] = 0u;
    g2[0] = 0u; g2[1] = 0u; g2[2] = 0u; g2[3] = 0u;
    g3[0] = 0u; g3[1] = 0u; g3[2] = 0u; g3[3] = 0u;
    asm volatile("tensor_load_to_lds %0, %1, %2, %3"
                 :: "s"(g0), "s"(g1), "s"(g2), "s"(g3)
                 : "memory");
    __builtin_amdgcn_s_wait_tensorcnt(0);
  }
  __syncthreads();

  v8f acc = {};
#pragma unroll
  for (int tap = 0; tap < 9; ++tap) {
    const int kh = tap / 3, kw = tap % 3;
    const _Float16* wbase = wt + tap * 4096 + (co0 + n) * 64;   // A row m = n
    const _Float16* srow  = smem + (kh * 18 + kw + n) * 64;     // lane's pixel
#pragma unroll
    for (int kc = 0; kc < 64; kc += 32) {
      v8h alo = *(const v8h*)(wbase + kc + halfid * 16);
      v8h ahi = *(const v8h*)(wbase + kc + halfid * 16 + 8);
      v8h blo = *(const v8h*)(srow + kc + halfid * 16);
      v8h bhi = *(const v8h*)(srow + kc + halfid * 16 + 8);
      acc = __builtin_amdgcn_wmma_f32_16x16x32_f16(false, cat8(alo, ahi),
                                                   false, cat8(blo, bhi),
                                                   (short)0, acc, false, false);
    }
  }
  if (mode == 0) {
    v8h o;
#pragma unroll
    for (int v = 0; v < 8; ++v) {
      float val = acc[v] + bias[co0 + halfid * 8 + v];
      o[v] = (_Float16)(val > 0.f ? val : 0.f);
    }
    // contiguous co range -> single 16B store into padded NHWC
    *(v8h*)&out_hp[(size_t)b * PIMG + (size_t)(h + 1) * PROW
                   + (size_t)(w + 1) * 64 + co0 + halfid * 8] = o;
  } else {
#pragma unroll
    for (int v = 0; v < 8; ++v) {
      const int co = co0 + halfid * 8 + v;
      const size_t idx = (((size_t)(b * 64 + co)) << 14) + (h << 7) + w;
      out_f[idx] = acc[v] + bias[co] + resid[idx];
    }
  }
}

// K3: per-pixel LayerNorm over 64 channels of NCHW conv_out -> token-major f16
__global__ void layernorm_tok(const float* __restrict__ co, const float* __restrict__ g,
                              const float* __restrict__ bta, _Float16* __restrict__ xn) {
  int tok = blockIdx.x * blockDim.x + threadIdx.x;
  if (tok >= NTOK) return;
  int b = tok >> 14, l = tok & 16383;
  const float* base = co + ((size_t)b << 20) + l;
  float vals[64], sum = 0.f;
#pragma unroll
  for (int c = 0; c < 64; ++c) { vals[c] = base[(size_t)c << 14]; sum += vals[c]; }
  float mu = sum * (1.f / 64.f), sq = 0.f;
#pragma unroll
  for (int c = 0; c < 64; ++c) { float d = vals[c] - mu; sq += d * d; }
  float rs = rsqrtf(sq * (1.f / 64.f) + 1e-5f);
#pragma unroll
  for (int c = 0; c < 64; ++c)
    xn[(size_t)tok * 64 + c] = (_Float16)((vals[c] - mu) * rs * g[c] + bta[c]);
}

// K4: in_proj  xz[N,256] = xn[N,64] @ W[256,64]^T    grid(2048,4) x 128thr
__global__ void gemm_inproj(const _Float16* __restrict__ X, const _Float16* __restrict__ Wm,
                            float* __restrict__ C) {
  const int lane = threadIdx.x & 31, wv = threadIdx.x >> 5;
  const size_t m0 = (size_t)blockIdx.x * 16;
  const int n0 = (blockIdx.y * 4 + wv) * 16;
  v8f acc = {};
  acc = wmma_tile_f16(X + m0 * 64, 64, Wm + (size_t)n0 * 64, 64, 64, acc);
  const int n = lane & 15, halfid = lane >> 4;
#pragma unroll
  for (int v = 0; v < 8; ++v)
    C[(m0 + halfid * 8 + v) * 256 + n0 + n] = acc[v];
}

// K5: depthwise causal conv1d (k=4) + bias + SiLU, xpart = xz[:, :128]
__global__ void conv1d_silu(const float* __restrict__ xz, const float* __restrict__ w,
                            const float* __restrict__ b, float* __restrict__ uf,
                            _Float16* __restrict__ uh) {
  int idx = blockIdx.x * blockDim.x + threadIdx.x;
  if (idx >= NTOK * 128) return;
  int d = idx & 127, l = (idx >> 7) & 16383, bb = idx >> 21;
  float acc = b[d];
#pragma unroll
  for (int k = 0; k < 4; ++k) {
    int ls = l - 3 + k;
    if (ls >= 0) acc += w[d * 4 + k] * xz[((size_t)(bb * 16384 + ls)) * 256 + d];
  }
  float s = acc / (1.f + __expf(-acc));
  uf[idx] = s;
  uh[idx] = (_Float16)s;
}

// K6: x_proj  dbl[N,48] = u[N,128] @ Wpad[48,128]^T   grid(2048,3) x 32thr
__global__ void gemm_xproj(const _Float16* __restrict__ U, const _Float16* __restrict__ Wm,
                           float* __restrict__ C) {
  const int lane = threadIdx.x & 31;
  const size_t m0 = (size_t)blockIdx.x * 16;
  const int n0 = blockIdx.y * 16;
  v8f acc = {};
  acc = wmma_tile_f16(U + m0 * 128, 128, Wm + (size_t)n0 * 128, 128, 128, acc);
  const int n = lane & 15, halfid = lane >> 4;
#pragma unroll
  for (int v = 0; v < 8; ++v)
    C[(m0 + halfid * 8 + v) * 48 + n0 + n] = acc[v];
}

// K7: dt = softplus(dt_r @ dtw^T + dtb), K = 4 (VALU is right here)
__global__ void dt_proj(const float* __restrict__ dbl, const float* __restrict__ w,
                        const float* __restrict__ b, float* __restrict__ dt) {
  int idx = blockIdx.x * blockDim.x + threadIdx.x;
  if (idx >= NTOK * 128) return;
  int d = idx & 127; size_t tok = (size_t)(idx >> 7);
  float a = b[d];
#pragma unroll
  for (int r = 0; r < 4; ++r) a += dbl[tok * 48 + r] * w[d * 4 + r];
  dt[idx] = (a > 20.f) ? a : log1pf(__expf(a));
}

// K8: selective scan. One wave = 2 channels of one batch (16 states per half-
// wave), butterfly-reduce h*C over s.  Prefetch next timesteps (latency-bound).
__global__ void sel_scan(const float* __restrict__ uf, const float* __restrict__ dtf,
                         const float* __restrict__ dbl, const float* __restrict__ xz,
                         const float* __restrict__ Aneg, const float* __restrict__ Dv,
                         _Float16* __restrict__ ys) {
  const int lane = threadIdx.x & 31;
  const int b  = blockIdx.x >> 6;
  const int d  = ((blockIdx.x & 63) << 1) + (lane >> 4);
  const int s  = lane & 15;
  const float Ad = Aneg[d * 16 + s];
  const float Dd = Dv[d];
  float h = 0.f;
  const size_t tokbase = (size_t)b << 14;
  for (int l = 0; l < 16384; ++l) {
    const size_t tok = tokbase + l;
    const int lp = (l + 64 < 16384) ? l + 64 : l;
    __builtin_prefetch(&uf [(tokbase + lp) * 128 + d], 0, 1);
    __builtin_prefetch(&dbl[(tokbase + lp) * 48 + s], 0, 1);
    const float uv  = uf [tok * 128 + d];
    const float dtv = dtf[tok * 128 + d];
    const float Bv  = dbl[tok * 48 + 4 + s];
    const float Cv  = dbl[tok * 48 + 20 + s];
    h = __expf(dtv * Ad) * h + (dtv * uv) * Bv;
    float p = h * Cv;
    p += __shfl_xor(p, 1); p += __shfl_xor(p, 2);
    p += __shfl_xor(p, 4); p += __shfl_xor(p, 8);
    if (s == 0) {
      float y  = p + uv * Dd;
      float zv = xz[tok * 256 + 128 + d];
      y *= zv / (1.f + __expf(-zv));          // * silu(z)
      ys[tok * 128 + d] = (_Float16)y;
    }
  }
}

// K9: out_proj + final combine  out = m * (conv_out + 1), NCHW scatter.
__global__ void gemm_out_final(const _Float16* __restrict__ Y, const _Float16* __restrict__ Wo,
                               const float* __restrict__ co, float* __restrict__ outp) {
  const int lane = threadIdx.x & 31, wv = threadIdx.x >> 5;
  const size_t m0 = (size_t)blockIdx.x * 16;
  const int n0 = wv * 16;
  v8f acc = {};
  acc = wmma_tile_f16(Y + m0 * 128, 128, Wo + (size_t)n0 * 128, 128, 128, acc);
  const int n = lane & 15, halfid = lane >> 4;
#pragma unroll
  for (int v = 0; v < 8; ++v) {
    const size_t tok = m0 + halfid * 8 + v;
    const int bb = (int)(tok >> 14), l = (int)(tok & 16383);
    const size_t gi = (((size_t)(bb * 64 + n0 + n)) << 14) + l;
    outp[gi] = acc[v] * (co[gi] + 1.f);
  }
}

// ---------------------------------------------------------------------------
extern "C" void kernel_launch(void* const* d_in, const int* in_sizes, int n_in,
                              void* d_out, int out_size, void* d_ws, size_t ws_size,
                              hipStream_t stream) {
  const float* x    = (const float*)d_in[0];
  const float* c1w  = (const float*)d_in[1];
  const float* c1b  = (const float*)d_in[2];
  const float* c2w  = (const float*)d_in[3];
  const float* c2b  = (const float*)d_in[4];
  const float* lng  = (const float*)d_in[5];
  const float* lnb  = (const float*)d_in[6];
  const float* ipw  = (const float*)d_in[7];
  const float* c1dw = (const float*)d_in[8];
  const float* c1db = (const float*)d_in[9];
  const float* xpw  = (const float*)d_in[10];
  const float* dtw  = (const float*)d_in[11];
  const float* dtb  = (const float*)d_in[12];
  const float* alog = (const float*)d_in[13];
  const float* Dv   = (const float*)d_in[14];
  const float* opw  = (const float*)d_in[15];
  float* outp = (float*)d_out;

  char* ws = (char*)d_ws;
  size_t off = 0;
  auto alloc = [&](size_t bytes) {
    char* p = ws + off;
    off += (bytes + 255) & ~(size_t)255;
    return p;
  };
  const size_t npad = (size_t)NB * PIMG;                // padded NHWC f16 elems
  _Float16* x_hp  = (_Float16*)alloc(npad * 2);
  _Float16* a1_hp = (_Float16*)alloc(npad * 2);
  _Float16* wt1   = (_Float16*)alloc(9 * 64 * 64 * 2);
  _Float16* wt2   = (_Float16*)alloc(9 * 64 * 64 * 2);
  _Float16* inw   = (_Float16*)alloc(256 * 64 * 2);
  _Float16* xpwh  = (_Float16*)alloc(48 * 128 * 2);
  _Float16* outw  = (_Float16*)alloc(64 * 128 * 2);
  float*    Aneg  = (float*)   alloc(128 * 16 * 4);
  float*    convo = (float*)   alloc((size_t)NB * 64 * HWSZ * 4);
  _Float16* xn_h  = (_Float16*)alloc((size_t)NTOK * 64 * 2);
  float*    xz    = (float*)   alloc((size_t)NTOK * 256 * 4);
  float*    u_f   = (float*)   alloc((size_t)NTOK * 128 * 4);
  _Float16* u_h   = (_Float16*)alloc((size_t)NTOK * 128 * 2);
  float*    dbl   = (float*)   alloc((size_t)NTOK * 48 * 4);
  float*    dt_f  = (float*)   alloc((size_t)NTOK * 128 * 4);
  _Float16* ys_h  = (_Float16*)alloc((size_t)NTOK * 128 * 2);
  (void)ws_size; (void)in_sizes; (void)n_in; (void)out_size;

  // 00: zero padded buffers (halo ring must be 0 for TDM tiles)
  zero_padded<<<((int)npad + 255) / 256, 256, 0, stream>>>(x_hp, a1_hp, (int)npad);
  // 0: weight packing + x cast into padded NHWC layout
  {
    int total = 9 * 64 * 64 + 256 * 64 + 48 * 128 + 64 * 128 + 128 * 16;
    pack_weights<<<(total + 255) / 256, 256, 0, stream>>>(
        c1w, c2w, ipw, xpw, opw, alog, wt1, wt2, inw, xpwh, outw, Aneg);
    int nx = NB * 64 * HWSZ;
    cast_to_padded<<<(nx + 255) / 256, 256, 0, stream>>>(x, x_hp);
  }
  // 1: conv1 + ReLU -> padded NHWC f16
  conv3x3_wmma<<<2048, 128, 0, stream>>>(x_hp, wt1, c1b, nullptr, nullptr, a1_hp, 0);
  // 2: conv2 + residual -> conv_out (f32, NCHW)
  conv3x3_wmma<<<2048, 128, 0, stream>>>(a1_hp, wt2, c2b, x, convo, nullptr, 1);
  // 3: LayerNorm over channels -> token-major f16
  layernorm_tok<<<NTOK / 256, 256, 0, stream>>>(convo, lng, lnb, xn_h);
  // 4: in_proj GEMM (N=256)
  gemm_inproj<<<dim3(NTOK / 16, 4), 128, 0, stream>>>(xn_h, inw, xz);
  // 5: depthwise conv1d + SiLU
  {
    int n = NTOK * 128;
    conv1d_silu<<<(n + 255) / 256, 256, 0, stream>>>(xz, c1dw, c1db, u_f, u_h);
  }
  // 6: x_proj GEMM (N padded to 48)
  gemm_xproj<<<dim3(NTOK / 16, 3), 32, 0, stream>>>(u_h, xpwh, dbl);
  // 7: dt_proj + softplus
  {
    int n = NTOK * 128;
    dt_proj<<<(n + 255) / 256, 256, 0, stream>>>(dbl, dtw, dtb, dt_f);
  }
  // 8: selective scan (latency-bound serial stage)
  sel_scan<<<NB * 64, 32, 0, stream>>>(u_f, dt_f, dbl, xz, Aneg, Dv, ys_h);
  // 9: out_proj GEMM + final combine
  gemm_out_final<<<NTOK / 16, 128, 0, stream>>>(ys_h, outw, convo, outp);
}
